// MOTMultiHeadSelfAttention_18537078850020
// MI455X (gfx1250) — matrix operand order repaired and checked by reading, not compiled, not run on hardware
//
#include <hip/hip_runtime.h>
#include <hip/hip_bf16.h>

// ---------- types / helpers ----------
typedef __attribute__((ext_vector_type(16))) __bf16 v16bf;
typedef __attribute__((ext_vector_type(8)))  float  v8f;

__device__ __forceinline__ v8f wmma_bf16(v16bf a, v16bf b, v8f c) {
  // D = A(16x32 bf16) * B(32x16 bf16) + C(16x16 f32)
  return __builtin_amdgcn_wmma_f32_16x16x32_bf16(false, a, false, b, (short)0, c,
                                                 false, false);
}

__device__ __forceinline__ __bf16 f2bf(float f) {
  unsigned u = __builtin_bit_cast(unsigned, f);
  u += 0x7FFFu + ((u >> 16) & 1u);               // round-to-nearest-even
  unsigned short h = (unsigned short)(u >> 16);
  return __builtin_bit_cast(__bf16, h);
}

// gfx1250 async global->LDS copy (16 bytes per lane), tracked by ASYNCcnt.
__device__ __forceinline__ void async_copy16B(unsigned lds_byte_off, const __bf16* g) {
  asm volatile("global_load_async_to_lds_b128 %0, %1, off"
               :: "v"(lds_byte_off), "v"(g) : "memory");
}
__device__ __forceinline__ void wait_async0() {
  asm volatile("s_wait_asynccnt 0x0" ::: "memory");
}
__device__ __forceinline__ unsigned lds_off(const void* p) {
  return (unsigned)(unsigned long long)p;  // low 32 bits of generic addr = LDS offset
}

// ---------- fp32 -> bf16 conversion (n is always a multiple of 1024) ----------
__global__ __launch_bounds__(256) void k_cvt_bf16(const float* __restrict__ src,
                                                  __bf16* __restrict__ dst,
                                                  long long n) {
  (void)n;
  long long i = ((long long)blockIdx.x * 256 + threadIdx.x) * 4;
#pragma unroll
  for (int j = 0; j < 4; ++j) dst[i + j] = f2bf(src[i + j]);
}

// ---------- generic bf16 WMMA GEMM:  C[M,N]f32 = A @ B + bias ----------
// A rows remapped: global row r -> b = r/Lmod, l = r%Lmod,
// row ptr = A + b*a_batch_stride + a_row_off + l*lda       (elements)
// M % 64 == 0, N % 64 == 0, K % 32 == 0 guaranteed by the launcher.
__global__ __launch_bounds__(128)
void k_gemm_bf16(const __bf16* __restrict__ A, long long a_batch_stride, int Lmod,
                 long long a_row_off, int lda,
                 const __bf16* __restrict__ Bm, int ldb,
                 const float* __restrict__ bias,
                 float* __restrict__ Cm, int ldc,
                 int M, int N, int K) {
  __shared__ __bf16 Asl[64 * 32];   // [m][k] row-major (memory layout copy)
  __shared__ __bf16 Bsl[64 * 32];   // [n][k] transposed -> contiguous frag reads
  const int tid  = threadIdx.x;
  const int wave = tid >> 5, lane = tid & 31;
  const int half = lane >> 4, l16 = lane & 15;
  const int m0 = (wave >> 1) * 32, n0 = (wave & 1) * 32;
  const int blockM = blockIdx.x * 64, blockN = blockIdx.y * 64;

  v8f acc[2][2];
#pragma unroll
  for (int mf = 0; mf < 2; ++mf)
#pragma unroll
    for (int nf = 0; nf < 2; ++nf) acc[mf][nf] = (v8f)0.0f;

  const int arow = tid >> 1, acol = (tid & 1) * 16;   // 64x32 A tile, 16 elem/thread
  const int brow = tid >> 2, bcol = (tid & 3) * 16;   // 32x64 B tile, 16 elem/thread
  const int gmA = blockM + arow;
  const int b_idx = gmA / Lmod, l_idx = gmA % Lmod;
  const __bf16* Arow = A + (long long)b_idx * a_batch_stride + a_row_off +
                       (long long)l_idx * lda;
  const unsigned aoff = lds_off(&Asl[arow * 32 + acol]);

  for (int kb = 0; kb < K; kb += 32) {
    __syncthreads();
    // A tile: async DMA straight into LDS (row-major copy)
    {
      const __bf16* g = Arow + kb + acol;
      async_copy16B(aoff, g);
      async_copy16B(aoff + 16u, g + 8);
    }
    // B tile: coalesced global read, transposed scatter into LDS
    {
      const uint4* s = (const uint4*)(Bm + (long long)(kb + brow) * ldb + blockN + bcol);
      uint4 r0 = s[0], r1 = s[1];
      __bf16 tmp[16];
      *(uint4*)(tmp + 0) = r0;
      *(uint4*)(tmp + 8) = r1;
#pragma unroll
      for (int i = 0; i < 16; ++i) Bsl[(bcol + i) * 32 + brow] = tmp[i];
    }
    if (kb + 32 < K)
      __builtin_prefetch(Bm + (long long)(kb + 32 + brow) * ldb + blockN + bcol, 0, 1);
    wait_async0();
    __syncthreads();

    v16bf afr[2], bfr[2];
#pragma unroll
    for (int mf = 0; mf < 2; ++mf) {
      const __bf16* ar = &Asl[(m0 + mf * 16 + l16) * 32];
#pragma unroll
      for (int e = 0; e < 8; ++e) {
        afr[mf][e]     = ar[8 * half + e];
        afr[mf][8 + e] = ar[16 + 8 * half + e];
      }
    }
#pragma unroll
    for (int nf = 0; nf < 2; ++nf) {
      const __bf16* br = &Bsl[(n0 + nf * 16 + l16) * 32];
#pragma unroll
      for (int e = 0; e < 16; ++e) bfr[nf][e] = br[e < 8 ? 8 * half + e : 16 + 8 * half + (e - 8)];   // 2x ds_load_b128
    }
#pragma unroll
    for (int mf = 0; mf < 2; ++mf)
#pragma unroll
      for (int nf = 0; nf < 2; ++nf)
        acc[mf][nf] = wmma_bf16(afr[mf], bfr[nf], acc[mf][nf]);
  }

#pragma unroll
  for (int mf = 0; mf < 2; ++mf)
#pragma unroll
    for (int nf = 0; nf < 2; ++nf) {
      const int gn = blockN + n0 + nf * 16 + l16;
      const float bv = bias[gn];
#pragma unroll
      for (int v = 0; v < 8; ++v) {
        const int gm = blockM + m0 + mf * 16 + v + 8 * half;
        Cm[(long long)gm * ldc + gn] = acc[mf][nf][v] + bv;
      }
    }
}

// ---------- RMS-norm q/k + pack q,k,v to unified bf16 [B,H,3584,64] ----------
__global__ __launch_bounds__(128)
void k_qkv_post(const float* __restrict__ qkv,  // [B, Lmod, 3, 16, 64]
                const float* __restrict__ gq, const float* __restrict__ gk,
                __bf16* __restrict__ Qb, __bf16* __restrict__ Kb,
                __bf16* __restrict__ Vb, int Lmod, int Loff) {
  const int wid  = blockIdx.x * 4 + (threadIdx.x >> 5);
  const int lane = threadIdx.x & 31;
  if (wid >= 2 * Lmod * 16) return;
  const int h = wid & 15;
  const int l = (wid >> 4) % Lmod;
  const int b = (wid >> 4) / Lmod;

  const float* base = qkv + ((((long long)b * Lmod + l) * 3) * 16 + h) * 64;
  const float q0 = base[lane],        q1 = base[lane + 32];
  const float k0 = base[1024 + lane], k1 = base[1024 + lane + 32];
  const float v0 = base[2048 + lane], v1 = base[2048 + lane + 32];

  float sq = q0 * q0 + q1 * q1;
  float sk = k0 * k0 + k1 * k1;
#pragma unroll
  for (int m = 16; m >= 1; m >>= 1) {
    sq += __shfl_xor(sq, m, 32);
    sk += __shfl_xor(sk, m, 32);
  }
  const float rq = 8.0f / fmaxf(sqrtf(sq), 1e-12f);  // SCALE_RMS = sqrt(64)
  const float rk = 8.0f / fmaxf(sqrtf(sk), 1e-12f);

  const long long orow = (((long long)b * 16 + h) * 3584 + (Loff + l)) * 64;
  const float* gqh = gq + h * 64;
  const float* gkh = gk + h * 64;
  Qb[orow + lane]      = f2bf(q0 * rq * gqh[lane]);
  Qb[orow + lane + 32] = f2bf(q1 * rq * gqh[lane + 32]);
  Kb[orow + lane]      = f2bf(k0 * rk * gkh[lane]);
  Kb[orow + lane + 32] = f2bf(k1 * rk * gkh[lane + 32]);
  Vb[orow + lane]      = f2bf(v0);
  Vb[orow + lane + 32] = f2bf(v1);
}

// ---------- flash attention: 1 wave = 32 q rows, kv chunks of 32 ----------
__global__ __launch_bounds__(128)
void k_attn(const __bf16* __restrict__ Qb, const __bf16* __restrict__ Kb,
            const __bf16* __restrict__ Vb,
            __bf16* __restrict__ Hb,  // [B, 3584, 16, 64]
            int q_off, int kv_off, int Lkv) {
  __shared__ __bf16 Ksl[32 * 64];      // [kv][d] row-major (memory copy, async)
  __shared__ __bf16 Vsl[64 * 32];      // [d][kv] transposed -> contiguous frag reads
  __shared__ __bf16 Psl[4][32 * 32];

  const int tid  = threadIdx.x;
  const int wave = tid >> 5, lane = tid & 31;
  const int half = lane >> 4, l16 = lane & 15;
  const int bh = blockIdx.y, b = bh >> 4, h = bh & 15;
  const int q0g = blockIdx.x * 128 + wave * 32;
  const long long baseQ  = (((long long)b * 16 + h) * 3584 + q_off + q0g) * 64;
  const long long baseKV = (((long long)b * 16 + h) * 3584 + kv_off) * 64;

  // preload Q fragments: [mfrag][kchunk]
  v16bf qfr[2][2];
#pragma unroll
  for (int mf = 0; mf < 2; ++mf)
#pragma unroll
    for (int kc = 0; kc < 2; ++kc) {
      const __bf16* qr = Qb + baseQ + (long long)(mf * 16 + l16) * 64 + kc * 32;
#pragma unroll
      for (int e = 0; e < 8; ++e) {
        qfr[mf][kc][e]     = qr[8 * half + e];
        qfr[mf][kc][8 + e] = qr[16 + 8 * half + e];
      }
    }

  v8f Ofr[2][4];
#pragma unroll
  for (int mf = 0; mf < 2; ++mf)
#pragma unroll
    for (int nf = 0; nf < 4; ++nf) Ofr[mf][nf] = (v8f)0.0f;
  float mrun[2][8], lrun[2][8];
#pragma unroll
  for (int mf = 0; mf < 2; ++mf)
#pragma unroll
    for (int v = 0; v < 8; ++v) { mrun[mf][v] = -1e30f; lrun[mf][v] = 0.0f; }

  const int krow = tid >> 2, kcol = (tid & 3) * 16;
  const unsigned koff = lds_off(&Ksl[krow * 64 + kcol]);

  for (int kv = 0; kv < Lkv; kv += 32) {
    __syncthreads();
    // K tile: async DMA into LDS (row-major copy)
    {
      const __bf16* g = Kb + baseKV + (long long)(kv + krow) * 64 + kcol;
      async_copy16B(koff, g);
      async_copy16B(koff + 16u, g + 8);
    }
    // V tile: coalesced read, transposed scatter into LDS
    {
      const uint4* s = (const uint4*)(Vb + baseKV + (long long)(kv + krow) * 64 + kcol);
      uint4 r0 = s[0], r1 = s[1];
      __bf16 tmp[16];
      *(uint4*)(tmp + 0) = r0;
      *(uint4*)(tmp + 8) = r1;
#pragma unroll
      for (int i = 0; i < 16; ++i) Vsl[(kcol + i) * 32 + krow] = tmp[i];
    }
    if (kv + 32 < Lkv) {
      __builtin_prefetch(Kb + baseKV + (long long)(kv + 32 + krow) * 64 + kcol, 0, 1);
      __builtin_prefetch(Vb + baseKV + (long long)(kv + 32 + krow) * 64 + kcol, 0, 1);
    }
    wait_async0();
    __syncthreads();

    // S = (Q K^T) * 1/sqrt(64)
    v8f S[2][2];
#pragma unroll
    for (int mf = 0; mf < 2; ++mf)
#pragma unroll
      for (int nf = 0; nf < 2; ++nf) S[mf][nf] = (v8f)0.0f;
#pragma unroll
    for (int nf = 0; nf < 2; ++nf)
#pragma unroll
      for (int kc = 0; kc < 2; ++kc) {
        v16bf bfr;  // B[k=d][n=kv_local] = K[kv_local][d] -> contiguous row read
        const __bf16* kr = &Ksl[(nf * 16 + l16) * 64 + kc * 32];
#pragma unroll
        for (int e = 0; e < 16; ++e) bfr[e] = kr[e < 8 ? 8 * half + e : 16 + 8 * half + (e - 8)];
#pragma unroll
        for (int mf = 0; mf < 2; ++mf)
          S[mf][nf] = wmma_bf16(qfr[mf][kc], bfr, S[mf][nf]);
      }
#pragma unroll
    for (int mf = 0; mf < 2; ++mf)
#pragma unroll
      for (int nf = 0; nf < 2; ++nf)
#pragma unroll
        for (int v = 0; v < 8; ++v) S[mf][nf][v] *= 0.125f;

    // online softmax per row (row = v + 8*half within 16-row frag)
#pragma unroll
    for (int mf = 0; mf < 2; ++mf) {
      float mc[8], mnew[8], scl[8], ls[8];
#pragma unroll
      for (int v = 0; v < 8; ++v) {
        mc[v] = fmaxf(S[mf][0][v], S[mf][1][v]);
#pragma unroll
        for (int m = 8; m >= 1; m >>= 1) mc[v] = fmaxf(mc[v], __shfl_xor(mc[v], m, 32));
        mnew[v] = fmaxf(mrun[mf][v], mc[v]);
        scl[v]  = __expf(mrun[mf][v] - mnew[v]);
        mrun[mf][v] = mnew[v];
      }
#pragma unroll
      for (int nf = 0; nf < 4; ++nf)
#pragma unroll
        for (int v = 0; v < 8; ++v) Ofr[mf][nf][v] *= scl[v];
#pragma unroll
      for (int v = 0; v < 8; ++v) ls[v] = 0.0f;
#pragma unroll
      for (int nf = 0; nf < 2; ++nf)
#pragma unroll
        for (int v = 0; v < 8; ++v) {
          const float p = __expf(S[mf][nf][v] - mnew[v]);
          Psl[wave][(mf * 16 + v + 8 * half) * 32 + nf * 16 + l16] = f2bf(p);
          ls[v] += p;
        }
#pragma unroll
      for (int v = 0; v < 8; ++v) {
#pragma unroll
        for (int m = 8; m >= 1; m >>= 1) ls[v] += __shfl_xor(ls[v], m, 32);
        lrun[mf][v] = lrun[mf][v] * scl[v] + ls[v];
      }
    }

    // O += P @ V   (same-wave LDS store->load is in-order)
#pragma unroll
    for (int mf = 0; mf < 2; ++mf) {
      v16bf pa;
      const __bf16* pr = &Psl[wave][(mf * 16 + l16) * 32];
#pragma unroll
      for (int e = 0; e < 8; ++e) {
        pa[e]     = pr[8 * half + e];
        pa[8 + e] = pr[16 + 8 * half + e];
      }
#pragma unroll
      for (int nf = 0; nf < 4; ++nf) {
        v16bf vb;
        const __bf16* vr = &Vsl[(nf * 16 + l16) * 32];
#pragma unroll
        for (int e = 0; e < 16; ++e) vb[e] = vr[e < 8 ? 8 * half + e : 16 + 8 * half + (e - 8)];     // 2x ds_load_b128
        Ofr[mf][nf] = wmma_bf16(pa, vb, Ofr[mf][nf]);
      }
    }
  }

  // finalize: O /= l, write [B, L, H, D] bf16
#pragma unroll
  for (int mf = 0; mf < 2; ++mf) {
    float inv[8];
#pragma unroll
    for (int v = 0; v < 8; ++v) inv[v] = 1.0f / fmaxf(lrun[mf][v], 1e-20f);
#pragma unroll
    for (int nf = 0; nf < 4; ++nf) {
      const int d = nf * 16 + l16;
#pragma unroll
      for (int v = 0; v < 8; ++v) {
        const int qrow = q0g + mf * 16 + v + 8 * half;
        const long long o = (((long long)b * 3584 + q_off + qrow) * 16 + h) * 64 + d;
        Hb[o] = f2bf(Ofr[mf][nf][v] * inv[v]);
      }
    }
  }
}

// ---------- host launch ----------
extern "C" void kernel_launch(void* const* d_in, const int* in_sizes, int n_in,
                              void* d_out, int out_size, void* d_ws, size_t ws_size,
                              hipStream_t stream) {
  (void)in_sizes; (void)n_in; (void)out_size; (void)ws_size;
  const int Bn = 2, C = 1024, N3 = 3072, Ltot = 3584;
  const int Lm[3]   = {2048, 1024, 512};   // shape, texture, color
  const int Loff[3] = {1536, 0, 1024};     // unified order: [texture|color|shape]

  const float* xs[3]   = {(const float*)d_in[0],  (const float*)d_in[7],  (const float*)d_in[14]};
  const float* Wqkv[3] = {(const float*)d_in[1],  (const float*)d_in[8],  (const float*)d_in[15]};
  const float* bqkv[3] = {(const float*)d_in[2],  (const float*)d_in[9],  (const float*)d_in[16]};
  const float* gq[3]   = {(const float*)d_in[3],  (const float*)d_in[10], (const float*)d_in[17]};
  const float* gk[3]   = {(const float*)d_in[4],  (const float*)d_in[11], (const float*)d_in[18]};
  const float* Wout[3] = {(const float*)d_in[5],  (const float*)d_in[12], (const float*)d_in[19]};
  const float* bout[3] = {(const float*)d_in[6],  (const float*)d_in[13], (const float*)d_in[20]};

  size_t woff = 0;
  auto alloc = [&](size_t bytes) -> void* {
    void* p = (char*)d_ws + woff;
    woff = (woff + bytes + 255) & ~(size_t)255;
    return p;
  };

  __bf16* xbf[3];  __bf16* Wqkvbf[3];  __bf16* Woutbf[3];  float* qkvf[3];
  for (int m = 0; m < 3; ++m) xbf[m]    = (__bf16*)alloc((size_t)Bn * Lm[m] * C * 2);
  for (int m = 0; m < 3; ++m) Wqkvbf[m] = (__bf16*)alloc((size_t)C * N3 * 2);
  for (int m = 0; m < 3; ++m) Woutbf[m] = (__bf16*)alloc((size_t)C * C * 2);
  for (int m = 0; m < 3; ++m) qkvf[m]   = (float*)alloc((size_t)Bn * Lm[m] * N3 * 4);
  __bf16* Qb = (__bf16*)alloc((size_t)Bn * 16 * Ltot * 64 * 2);
  __bf16* Kb = (__bf16*)alloc((size_t)Bn * 16 * Ltot * 64 * 2);
  __bf16* Vb = (__bf16*)alloc((size_t)Bn * 16 * Ltot * 64 * 2);
  __bf16* Hb = (__bf16*)alloc((size_t)Bn * Ltot * C * 2);

  // 1) fp32 -> bf16 conversions (all sizes are multiples of 1024)
  for (int m = 0; m < 3; ++m) {
    long long n = (long long)Bn * Lm[m] * C;
    k_cvt_bf16<<<(unsigned)(n / 1024), 256, 0, stream>>>(xs[m], xbf[m], n);
    long long nq = (long long)C * N3;
    k_cvt_bf16<<<(unsigned)(nq / 1024), 256, 0, stream>>>(Wqkv[m], Wqkvbf[m], nq);
    long long no = (long long)C * C;
    k_cvt_bf16<<<(unsigned)(no / 1024), 256, 0, stream>>>(Wout[m], Woutbf[m], no);
  }

  // 2) QKV projection GEMMs (M = B*Lmod, N = 3072, K = 1024)
  for (int m = 0; m < 3; ++m) {
    const int M = Bn * Lm[m];
    dim3 grid(M / 64, N3 / 64);
    k_gemm_bf16<<<grid, 128, 0, stream>>>(xbf[m], 0LL, M, 0LL, C,
                                          Wqkvbf[m], N3, bqkv[m],
                                          qkvf[m], N3, M, N3, C);
  }

  // 3) RMS norm + pack into unified bf16 Q/K/V
  for (int m = 0; m < 3; ++m) {
    const int rows = Bn * Lm[m] * 16;
    k_qkv_post<<<rows / 4, 128, 0, stream>>>(qkvf[m], gq[m], gk[m],
                                             Qb, Kb, Vb, Lm[m], Loff[m]);
  }

  // 4) attention: shape self-attn, then texture+color cross-attn
  k_attn<<<dim3(2048 / 128, Bn * 16), 128, 0, stream>>>(Qb, Kb, Vb, Hb, 1536, 1536, 2048);
  k_attn<<<dim3(1536 / 128, Bn * 16), 128, 0, stream>>>(Qb, Kb, Vb, Hb, 0, 0, 3584);

  // 5) output projections straight into d_out (shape, texture, color)
  long long ooff = 0;
  for (int m = 0; m < 3; ++m) {
    const int M = Bn * Lm[m];
    dim3 grid(M / 64, C / 64);
    k_gemm_bf16<<<grid, 128, 0, stream>>>(Hb, (long long)Ltot * C, Lm[m],
                                          (long long)Loff[m] * C, C,
                                          Woutbf[m], C, bout[m],
                                          (float*)d_out + ooff, C, M, C, C);
    ooff += (long long)M * C;
  }
}
